// LinearCrossAttention_54760833024196
// MI455X (gfx1250) — matrix operand order repaired and checked by reading, not compiled, not run on hardware
//
#include <hip/hip_runtime.h>
#include <hip/hip_bf16.h>

typedef __attribute__((ext_vector_type(16))) __bf16 v16bf;
typedef __attribute__((ext_vector_type(8)))  float  v8f;
typedef __attribute__((ext_vector_type(4)))  float  v4f;

#define EPS_F 1e-6f

__device__ __forceinline__ float wave_sum(float x) {
  #pragma unroll
  for (int off = 16; off > 0; off >>= 1) x += __shfl_xor(x, off, 32);
  return x;
}

__device__ __forceinline__ float sigmoid_f(float x) {
  return 1.0f / (1.0f + expf(-x));
}

// fp32 -> bf16 (RNE) weight pre-conversion
__global__ void cvt_bf16(const float* __restrict__ src, __bf16* __restrict__ dst, int n) {
  int i = blockIdx.x * blockDim.x + threadIdx.x;
  if (i < n) dst[i] = (__bf16)src[i];
}

// Y[r][c] = sum_k X[r][k] * W[c][k] + bias[c] (+pos[r][c])
// If remapN>0: out row is remapped (r = b*remapN+n -> n*remapB+b) and residual added.
// One 16(M)x64(N) output strip per wave: the A fragment is built once per
// K-step and reused by 4 independent v_wmma_f32_16x16x32_bf16 accumulators.
// R rows (multiple of 32); tile-groups = (R/16)*4 = R/4; 8 waves/block.
__global__ __launch_bounds__(256) void gemm_wmma(
    const float* __restrict__ X, const __bf16* __restrict__ W,
    const float* __restrict__ bias, const float* __restrict__ pos,
    const float* __restrict__ residual, float* __restrict__ Y,
    int R, int remapN, int remapB)
{
  const int lane = threadIdx.x & 31;
  const int wave = threadIdx.x >> 5;
  const int tg   = blockIdx.x * 8 + wave;     // wave-uniform -> uniform early exit
  if (tg >= (R >> 2)) return;
  const int mbase = (tg >> 2) << 4;           // 16-row block
  const int nbase = (tg & 3) << 6;            // 64-col strip
  const int half  = lane >> 4;                // 0: lanes 0-15, 1: lanes 16-31
  const int lid   = lane & 15;

  const float*  xrow  = X + (size_t)(mbase + lid) * 256;            // A row M
  const __bf16* wbase = W + (size_t)(nbase + lid) * 256 + half * 16; // B col N (j=0)

  v8f acc[4] = {};
  #pragma unroll
  for (int k0 = 0; k0 < 256; k0 += 32) {
    // A fragment (contiguous K order, matching B): lane<16 holds K=0..15+k0,
    // lane>=16 holds K=16..31+k0 of row M = lane&15.
    const float* p0 = xrow + k0 + half * 16;
    const float* p1 = xrow + k0 + half * 16 + 8;
    const v4f x0 = *(const v4f*)(p0);
    const v4f x1 = *(const v4f*)(p0 + 4);
    const v4f x2 = *(const v4f*)(p1);
    const v4f x3 = *(const v4f*)(p1 + 4);
    v16bf a;
    #pragma unroll
    for (int e = 0; e < 4; ++e) {
      a[e]      = (__bf16)x0[e];
      a[e + 4]  = (__bf16)x1[e];
      a[e + 8]  = (__bf16)x2[e];
      a[e + 12] = (__bf16)x3[e];
    }
    // 4 B fragments (32x16 each): lane&15 = N column, half selects K 0..15 /
    // 16..31; 16 contiguous bf16 (32B aligned). 4 independent accumulators
    // hide the WMMA D-write hazard.
    #pragma unroll
    for (int j = 0; j < 4; ++j) {
      v16bf b = *(const v16bf*)(wbase + (size_t)j * 16 * 256 + k0);
      acc[j] = __builtin_amdgcn_wmma_f32_16x16x32_bf16(false, a, false, b,
                                                       (short)0, acc[j], false, false);
    }
  }

  #pragma unroll
  for (int j = 0; j < 4; ++j) {
    const int col = nbase + j * 16 + lid;
    const float bc = bias[col];
    #pragma unroll
    for (int r = 0; r < 8; ++r) {
      const int row = mbase + half * 8 + r;   // D layout: VGPR r -> M = r (+8 hi half)
      float v = acc[j][r] + bc;
      if (remapN > 0) {
        const int b_ = row / remapN;
        const int n_ = row - b_ * remapN;
        const size_t o = (size_t)(n_ * remapB + b_) * 256 + col;
        Y[o] = v + residual[o];
      } else {
        const size_t o = (size_t)row * 256 + col;
        if (pos) v += pos[o];
        Y[o] = v;
      }
    }
  }
}

// One (b,l) position per wave; lane = d (D==32==wave width).
// All cumsums run over the head axis (H=8) exactly as in the reference.
// Writes att[b][h][l][d] so that a flat (B*N,256) view of the buffer matches
// the reference's swapaxes(1,2).reshape(B,N,C).
__global__ __launch_bounds__(256) void attn_kernel(
    const float* __restrict__ qb, const float* __restrict__ kb,
    const float* __restrict__ vb, float* __restrict__ att,
    int Nq, int Bsz)
{
  const int wave = blockIdx.x * 8 + (threadIdx.x >> 5);
  if (wave >= Nq * Bsz) return;               // wave-uniform
  const int lane = threadIdx.x & 31;
  const int b = wave % Bsz;
  const int l = wave / Bsz;
  const size_t row = ((size_t)l * Bsz + b) * 256;

  float q[8], k[8], v[8];
  #pragma unroll
  for (int h = 0; h < 8; ++h) {
    q[h] = sigmoid_f(qb[row + h * 32 + lane]);
    k[h] = sigmoid_f(kb[row + h * 32 + lane]);
    v[h] = vb[row + h * 32 + lane];
  }

  float sink_in[8], src_out[8];
  {
    float ck = 0.f, cq = 0.f;
    #pragma unroll
    for (int h = 0; h < 8; ++h) {
      ck += k[h]; cq += q[h];
      const float nm = (float)(h + 1);
      sink_in[h] = nm / wave_sum((q[h] + EPS_F) * (ck + EPS_F));
      src_out[h] = nm / wave_sum((k[h] + EPS_F) * (cq + EPS_F));
    }
  }

  float sink_alloc[8], src_comp[8];
  {
    float ckso = 0.f, cqsi = 0.f, ccs = 0.f;
    #pragma unroll
    for (int h = 0; h < 8; ++h) {
      ckso += k[h] * src_out[h];
      cqsi += q[h] * sink_in[h];
      const float nm = (float)(h + 1);
      const float csk = wave_sum((q[h] + EPS_F) * (ckso + EPS_F)) / nm;
      float csr = wave_sum((k[h] + EPS_F) * (cqsi + EPS_F)) / nm;
      csr = fminf(1.0f, fmaxf(-1.0f, csr));
      sink_alloc[h] = sigmoid_f(csk);
      const float e = expf(csr);
      ccs += e;
      src_comp[h] = e / ccs * nm;
    }
  }

  float qs[8], vs[8], acc[8];
  #pragma unroll
  for (int h = 0; h < 8; ++h) {
    qs[h]  = q[h] * sink_in[h] / (float)(h + 1);
    vs[h]  = v[h] * src_comp[h];
    acc[h] = 0.f;
  }
  // qkv[h][m] = sum_{h'<=h} <qs[h], k[h']> * vs[h'][m]
  #pragma unroll
  for (int hp = 0; hp < 8; ++hp) {
    #pragma unroll
    for (int h = hp; h < 8; ++h) {
      const float S = wave_sum(qs[h] * k[hp]);
      acc[h] += S * vs[hp];
    }
  }

  const size_t base = (size_t)b * Nq * 256 + (size_t)l * 32 + lane;
  #pragma unroll
  for (int h = 0; h < 8; ++h)
    att[base + (size_t)h * Nq * 32] = acc[h] * sink_alloc[h];
}

extern "C" void kernel_launch(void* const* d_in, const int* in_sizes, int n_in,
                              void* d_out, int out_size, void* d_ws, size_t ws_size,
                              hipStream_t stream)
{
  (void)in_sizes; (void)n_in; (void)out_size; (void)ws_size;
  const int N = 900, B = 8, C = 256;
  const int RQ = N * B;   // 7200 rows; l >= N never influences the output

  const float* query     = (const float*)d_in[0];
  const float* key       = (const float*)d_in[1];
  const float* query_pos = (const float*)d_in[2];
  const float* key_pos   = (const float*)d_in[3];
  const float* Wq = (const float*)d_in[4];  const float* bq = (const float*)d_in[5];
  const float* Wk = (const float*)d_in[6];  const float* bk = (const float*)d_in[7];
  const float* Wv = (const float*)d_in[8];  const float* bv = (const float*)d_in[9];
  const float* Wo = (const float*)d_in[10]; const float* bo = (const float*)d_in[11];

  char* ws = (char*)d_ws;
  __bf16* WqB = (__bf16*)ws;
  __bf16* WkB = WqB + 65536;
  __bf16* WvB = WkB + 65536;
  __bf16* WoB = WvB + 65536;
  float* qbuf = (float*)(ws + (size_t)4 * 65536 * sizeof(__bf16));
  float* kbuf = qbuf + (size_t)RQ * C;
  float* vbuf = kbuf + (size_t)RQ * C;
  float* attb = vbuf + (size_t)RQ * C;   // total ws use ~30 MB

  cvt_bf16<<<256, 256, 0, stream>>>(Wq, WqB, 65536);
  cvt_bf16<<<256, 256, 0, stream>>>(Wk, WkB, 65536);
  cvt_bf16<<<256, 256, 0, stream>>>(Wv, WvB, 65536);
  cvt_bf16<<<256, 256, 0, stream>>>(Wo, WoB, 65536);

  // Projections (only the first N sequence positions of key matter: the
  // attention body has no cross-position coupling and the tail is sliced off).
  gemm_wmma<<<RQ / 32, 256, 0, stream>>>(query, WqB, bq, query_pos, nullptr, qbuf, RQ, 0, 0);
  gemm_wmma<<<RQ / 32, 256, 0, stream>>>(key,   WkB, bk, key_pos,   nullptr, kbuf, RQ, 0, 0);
  gemm_wmma<<<RQ / 32, 256, 0, stream>>>(key,   WvB, bv, nullptr,   nullptr, vbuf, RQ, 0, 0);

  attn_kernel<<<RQ / 8, 256, 0, stream>>>(qbuf, kbuf, vbuf, attb, N, B);

  // Output projection + (b,n)->(n,b) remap + residual add of query.
  gemm_wmma<<<RQ / 32, 256, 0, stream>>>(attb, WoB, bo, nullptr, query, (float*)d_out, RQ, N, B);
}
